// SlotAttention_56865366999335
// MI455X (gfx1250) — matrix operand order repaired and checked by reading, not compiled, not run on hardware
//
#include <hip/hip_runtime.h>
#include <hip/hip_bf16.h>
#include <math.h>

#define KB 16
#define KN 4096
#define KS 16
#define KDIN 256
#define KDS 128
#define KDH 256
#define KR (KB*KN)
#define LN16 2.7725887222397812f

typedef __attribute__((ext_vector_type(16))) _Float16 v16h;
typedef __attribute__((ext_vector_type(8)))  float    v8f;

// ---------------------------------------------------------------- utilities

__device__ __forceinline__ float gauss_hash(unsigned s) {
  unsigned h = s;
  h ^= h >> 16; h *= 0x7feb352du; h ^= h >> 15; h *= 0x846ca68bu; h ^= h >> 16;
  unsigned h2 = h ^ 0x9e3779b9u;
  h2 ^= h2 >> 16; h2 *= 0x7feb352du; h2 ^= h2 >> 15; h2 *= 0x846ca68bu; h2 ^= h2 >> 16;
  float u1 = (h >> 8) * 5.9604645e-8f + 1e-12f;
  float u2 = (h2 >> 8) * 5.9604645e-8f;
  return sqrtf(-2.f * __logf(u1)) * cosf(6.28318530718f * u2);
}

__device__ __forceinline__ void loadRow16(const float* base, float* rp) {
  const float4* p4 = (const float4*)base;
  float4 a = p4[0], b = p4[1], c = p4[2], d = p4[3];
  rp[0]=a.x; rp[1]=a.y; rp[2]=a.z;  rp[3]=a.w;
  rp[4]=b.x; rp[5]=b.y; rp[6]=b.z;  rp[7]=b.w;
  rp[8]=c.x; rp[9]=c.y; rp[10]=c.z; rp[11]=c.w;
  rp[12]=d.x; rp[13]=d.y; rp[14]=d.z; rp[15]=d.w;
}

// 16-column (max,sum) logsumexp merge across a 1024-thread block (32 wave32s)
__device__ __forceinline__ void blockColLSE(float* cm, float* cs, float* Mt, float* St,
                                            float (*redm)[16], float (*reds)[16], int tid) {
  #pragma unroll
  for (int off = 16; off >= 1; off >>= 1) {
    #pragma unroll
    for (int c = 0; c < 16; ++c) {
      float om = __shfl_xor(cm[c], off, 32);
      float os = __shfl_xor(cs[c], off, 32);
      float M = fmaxf(cm[c], om);
      cs[c] = cs[c] * __expf(cm[c] - M) + os * __expf(om - M);
      cm[c] = M;
    }
  }
  int wv = tid >> 5;
  if ((tid & 31) == 0) {
    #pragma unroll
    for (int c = 0; c < 16; ++c) { redm[wv][c] = cm[c]; reds[wv][c] = cs[c]; }
  }
  __syncthreads();
  if (tid < 16) {
    float M = redm[0][tid], Ss = reds[0][tid];
    for (int w = 1; w < 32; ++w) {
      float om = redm[w][tid], os = reds[w][tid];
      float MM = fmaxf(M, om);
      Ss = Ss * __expf(M - MM) + os * __expf(om - MM);
      M = MM;
    }
    Mt[tid] = M; St[tid] = Ss;
  }
  __syncthreads();
}

// 16-column plain sum across 1024-thread block; result broadcast in out[16]
__device__ __forceinline__ void blockColSum(float* cv, float* out,
                                            float (*red)[16], int tid) {
  #pragma unroll
  for (int off = 16; off >= 1; off >>= 1) {
    #pragma unroll
    for (int c = 0; c < 16; ++c) cv[c] += __shfl_xor(cv[c], off, 32);
  }
  int wv = tid >> 5;
  if ((tid & 31) == 0) {
    #pragma unroll
    for (int c = 0; c < 16; ++c) red[wv][c] = cv[c];
  }
  __syncthreads();
  if (tid < 16) {
    float s = 0.f;
    for (int w = 0; w < 32; ++w) s += red[w][tid];
    out[tid] = s;
  }
  __syncthreads();
}

__device__ __forceinline__ float blockSum1024(float v, float* red32, int tid) {
  #pragma unroll
  for (int off = 16; off >= 1; off >>= 1) v += __shfl_xor(v, off, 32);
  if ((tid & 31) == 0) red32[tid >> 5] = v;
  __syncthreads();
  if (tid == 0) {
    float s = 0.f;
    for (int w = 0; w < 32; ++w) s += red32[w];
    red32[0] = s;
  }
  __syncthreads();
  float r = red32[0];
  __syncthreads();
  return r;
}

// ---------------------------------------------------------------- prep kernels

// wpT[col j][kpair p] so a lane's 8 B-matrix dwords are contiguous (2x b128)
__global__ void pack_w_kernel(const float* __restrict__ Wk, const float* __restrict__ Wv,
                              unsigned* __restrict__ wpT) {
  int p = blockIdx.x;          // K pair 0..127
  int j = threadIdx.x;         // output column 0..255
  float f0, f1;
  if (j < 128) { f0 = Wk[j * KDIN + 2 * p]; f1 = Wk[j * KDIN + 2 * p + 1]; }
  else         { f0 = Wv[(j - 128) * KDIN + 2 * p]; f1 = Wv[(j - 128) * KDIN + 2 * p + 1]; }
  union { unsigned u; _Float16 h[2]; } t;
  t.h[0] = (_Float16)f0; t.h[1] = (_Float16)f1;
  wpT[j * 128 + p] = t.u;
}

__global__ void prep_ln_kernel(const float* __restrict__ inp, const float* __restrict__ g,
                               const float* __restrict__ be, const float* __restrict__ Wa,
                               const float* __restrict__ ba,
                               _Float16* __restrict__ x16, float* __restrict__ alogit) {
  __shared__ float red[256];
  int row = blockIdx.x, t = threadIdx.x;
  float v = inp[(size_t)row * 256 + t];
  red[t] = v; __syncthreads();
  for (int o = 128; o >= 1; o >>= 1) { if (t < o) red[t] += red[t + o]; __syncthreads(); }
  float mean = red[0] * (1.f / 256.f); __syncthreads();
  float d = v - mean;
  red[t] = d * d; __syncthreads();
  for (int o = 128; o >= 1; o >>= 1) { if (t < o) red[t] += red[t + o]; __syncthreads(); }
  float var = red[0] * (1.f / 256.f); __syncthreads();
  float xln = d * rsqrtf(var + 1e-5f) * g[t] + be[t];
  x16[(size_t)row * 256 + t] = (_Float16)xln;
  red[t] = xln * Wa[t]; __syncthreads();
  for (int o = 128; o >= 1; o >>= 1) { if (t < o) red[t] += red[t + o]; __syncthreads(); }
  if (t == 0) alogit[row] = red[0] + ba[0];
}

// ---------------------------------------------------------------- WMMA GEMM: kv = LN(x) @ [Wk|Wv]^T

__global__ void gemm_xw_kernel(const unsigned* __restrict__ x16w,   // [R,128] dword pairs
                               const unsigned* __restrict__ wpT,    // [256 cols][128 pairs]
                               float* __restrict__ kv) {             // [R,256]
  int wave = threadIdx.x >> 5;
  int lane = threadIdx.x & 31;
  int m0 = blockIdx.x * 16;
  int n0 = (blockIdx.y * 8 + wave) * 16;
  int half = lane >> 4, mr = lane & 15;
  const unsigned* arow = x16w + (size_t)(m0 + mr) * 128;
  const unsigned* brow = wpT + (size_t)(n0 + mr) * 128;
  v8f c = {0.f, 0.f, 0.f, 0.f, 0.f, 0.f, 0.f, 0.f};
  for (int k0 = 0; k0 < 256; k0 += 32) {
    __builtin_prefetch(arow + ((k0 + 32) >> 1), 0, 1);
    union { v16h v; unsigned w[8]; } a, b;
    #pragma unroll
    for (int p = 0; p < 8; ++p) {
      int pa = (p < 4) ? (p + 4 * half) : (8 + (p - 4) + 4 * half);
      a.w[p] = arow[(k0 >> 1) + pa];
      b.w[p] = brow[(k0 >> 1) + pa];
    }
    c = __builtin_amdgcn_wmma_f32_16x16x32_f16(false, a.v, false, b.v, (short)0, c, false, false);
  }
  #pragma unroll
  for (int r = 0; r < 8; ++r)
    kv[(size_t)(m0 + r + 8 * half) * 256 + n0 + mr] = c[r];
}

__global__ void rownorm_kernel(const float* __restrict__ kv, float* __restrict__ kk) {
  int row = blockIdx.x * 256 + threadIdx.x;
  float s = 0.f;
  for (int d = 0; d < 128; ++d) { float x = kv[(size_t)row * 256 + d]; s += x * x; }
  kk[row] = s;
}

// val (loop-invariant) packed as [b][d][npair]: word = (val[2np][d], val[2np+1][d])
__global__ void pack_val_kernel(const float* __restrict__ kv, unsigned* __restrict__ vpT) {
  size_t idx = (size_t)blockIdx.x * 256 + threadIdx.x;   // b*128*2048 + d*2048 + np
  int np = (int)(idx & 2047);
  int d  = (int)((idx >> 11) & 127);
  int b  = (int)(idx >> 18);
  float f0 = kv[((size_t)(b * KN + 2 * np)) * 256 + 128 + d];
  float f1 = kv[((size_t)(b * KN + 2 * np + 1)) * 256 + 128 + d];
  union { unsigned u; _Float16 h[2]; } w;
  w.h[0] = (_Float16)f0; w.h[1] = (_Float16)f1;
  vpT[idx] = w.u;
}

__global__ void softmax_loga_kernel(const float* __restrict__ alogit, float* __restrict__ log_a) {
  __shared__ float red32[32];
  int b = blockIdx.x, t = threadIdx.x;
  float vv[4]; float m = -1e30f;
  #pragma unroll
  for (int r = 0; r < 4; ++r) { vv[r] = alogit[b * KN + t + 1024 * r]; m = fmaxf(m, vv[r]); }
  #pragma unroll
  for (int off = 16; off >= 1; off >>= 1) m = fmaxf(m, __shfl_xor(m, off, 32));
  if ((t & 31) == 0) red32[t >> 5] = m;
  __syncthreads();
  if (t == 0) { float mm = -1e30f; for (int w = 0; w < 32; ++w) mm = fmaxf(mm, red32[w]); red32[0] = mm; }
  __syncthreads();
  m = red32[0];
  __syncthreads();
  float s = 0.f;
  #pragma unroll
  for (int r = 0; r < 4; ++r) s += __expf(vv[r] - m);
  s = blockSum1024(s, red32, t);
  float lse = m + __logf(s);
  #pragma unroll
  for (int r = 0; r < 4; ++r) log_a[b * KN + t + 1024 * r] = vv[r] - lse + LN16;
}

__global__ void copy_kernel(const float* __restrict__ src, float* __restrict__ dst, int n) {
  int i = blockIdx.x * 256 + threadIdx.x;
  if (i < n) dst[i] = src[i];
}

__global__ void zero_kernel(float* __restrict__ dst, int n) {
  int i = blockIdx.x * 256 + threadIdx.x;
  if (i < n) dst[i] = 0.f;
}

// ---------------------------------------------------------------- per-iteration slot kernels

__global__ void slot_ln_kernel(const float* __restrict__ slots, const float* __restrict__ g,
                               const float* __restrict__ be, const float* __restrict__ Wb,
                               const float* __restrict__ bb, const float* __restrict__ Wq,
                               float* __restrict__ q, float* __restrict__ blogit,
                               float* __restrict__ qq) {
  __shared__ float red[128], snv[128];
  int bs = blockIdx.x, t = threadIdx.x;
  float x = slots[bs * KDS + t];
  red[t] = x; __syncthreads();
  for (int o = 64; o >= 1; o >>= 1) { if (t < o) red[t] += red[t + o]; __syncthreads(); }
  float mean = red[0] * (1.f / 128.f); __syncthreads();
  float d = x - mean;
  red[t] = d * d; __syncthreads();
  for (int o = 64; o >= 1; o >>= 1) { if (t < o) red[t] += red[t + o]; __syncthreads(); }
  float var = red[0] * (1.f / 128.f); __syncthreads();
  float sn = d * rsqrtf(var + 1e-5f) * g[t] + be[t];
  snv[t] = sn;
  red[t] = sn * Wb[t]; __syncthreads();
  for (int o = 64; o >= 1; o >>= 1) { if (t < o) red[t] += red[t + o]; __syncthreads(); }
  if (t == 0) blogit[bs] = red[0] + bb[0];
  __syncthreads();
  float acc = 0.f;
  for (int k2 = 0; k2 < 128; ++k2) acc += snv[k2] * Wq[t * KDS + k2];
  q[bs * KDS + t] = acc;
  red[t] = acc * acc; __syncthreads();
  for (int o = 64; o >= 1; o >>= 1) { if (t < o) red[t] += red[t + o]; __syncthreads(); }
  if (t == 0) qq[bs] = red[0];
}

// qpackT[b][slot n][kpair p]: contiguous per lane for B-fragment b128 loads
__global__ void slot_small_kernel(const float* __restrict__ blogit, const float* __restrict__ q,
                                  float* __restrict__ log_b, unsigned* __restrict__ qpackT) {
  int b = blockIdx.x, t = threadIdx.x;
  if (t < 16) {
    float m = -1e30f;
    for (int j = 0; j < 16; ++j) m = fmaxf(m, blogit[b * 16 + j]);
    float s = 0.f;
    for (int j = 0; j < 16; ++j) s += __expf(blogit[b * 16 + j] - m);
    log_b[b * 16 + t] = blogit[b * 16 + t] - (m + __logf(s)) + LN16;
  }
  int n = t >> 6, p = t & 63;
  float f0 = q[(b * 16 + n) * KDS + 2 * p];
  float f1 = q[(b * 16 + n) * KDS + 2 * p + 1];
  union { unsigned u; _Float16 h[2]; } w;
  w.h[0] = (_Float16)f0; w.h[1] = (_Float16)f1;
  qpackT[(b * 16 + n) * 64 + p] = w.u;
}

// ---------------------------------------------------------------- WMMA cdist: C_t = sqrt(||k||^2+||q||^2-2 k.q) + 1e-3*noise

__global__ void cdist_kernel(const float* __restrict__ kv, const unsigned* __restrict__ qpackT,
                             const float* __restrict__ kk, const float* __restrict__ qq,
                             float* __restrict__ Ct, int iter) {
  int lane = threadIdx.x & 31;
  int b = blockIdx.y, m0 = blockIdx.x * 16;
  int half = lane >> 4, mr = lane & 15;
  int rowbase = b * KN + m0;
  const float* arow = kv + (size_t)(rowbase + mr) * 256;
  const unsigned* brow = qpackT + (size_t)(b * 16 + mr) * 64;
  v8f c = {0.f, 0.f, 0.f, 0.f, 0.f, 0.f, 0.f, 0.f};
  for (int k0 = 0; k0 < 128; k0 += 32) {
    union { v16h v; _Float16 h[16]; } a;
    union { v16h v; unsigned w[8]; } bb_;
    #pragma unroll
    for (int p = 0; p < 8; ++p) {
      int ka = (p < 4) ? (2 * p + 8 * half) : (16 + 2 * (p - 4) + 8 * half);
      float f0 = arow[k0 + ka];
      float f1 = arow[k0 + ka + 1];
      a.h[2 * p] = (_Float16)f0; a.h[2 * p + 1] = (_Float16)f1;
      bb_.w[p] = brow[(k0 >> 1) + (ka >> 1)];
    }
    c = __builtin_amdgcn_wmma_f32_16x16x32_f16(false, a.v, false, bb_.v, (short)0, c, false, false);
  }
  int n = mr;
  #pragma unroll
  for (int r = 0; r < 8; ++r) {
    int m = m0 + r + 8 * half;
    float dot = c[r];
    float sq = kk[b * KN + m] + qq[b * 16 + n] - 2.f * dot;
    float c0 = sqrtf(fmaxf(sq, 0.f));
    unsigned seed = (unsigned)(((b * KN + m) * 16 + n) * 7 + iter * 0x68bc21ebu + 0x1234567u);
    Ct[(size_t)(b * KN + m) * 16 + n] = c0 + 1e-3f * gauss_hash(seed);
  }
}

__global__ void zero_uv_kernel(float* __restrict__ u, float* __restrict__ v) {
  int i = blockIdx.x * 256 + threadIdx.x;
  u[i] = 0.f;
  if (i < KB * KS) v[i] = 0.f;
}

// ---------------------------------------------------------------- MESH step: sinkhorn fwd + exact entropy grad + C update

__global__ void mesh_step_kernel(float* __restrict__ Ct, float* __restrict__ u,
                                 float* __restrict__ v, const float* __restrict__ log_a,
                                 const float* __restrict__ log_b) {
  __shared__ float redm[32][16], reds[32][16];
  __shared__ float vcur[16], vh[6][16], lbs[16], Mt[16], St[16], gvs[16], red32[32];
  int b = blockIdx.x, tid = threadIdx.x;
  float* C = Ct + (size_t)b * KN * 16;
  float ur[4], la[4];
  #pragma unroll
  for (int r = 0; r < 4; ++r) {
    ur[r] = u[b * KN + tid + 1024 * r];
    la[r] = log_a[b * KN + tid + 1024 * r];
  }
  if (tid < 16) {
    float vi = v[b * 16 + tid];
    vcur[tid] = vi; vh[0][tid] = vi;
    lbs[tid] = log_b[b * 16 + tid];
  }
  __syncthreads();

  // ---- forward: 5 sinkhorn iterations, storing v history
  for (int k = 1; k <= 5; ++k) {
    float cm[16], cs[16];
    #pragma unroll
    for (int c = 0; c < 16; ++c) { cm[c] = -1e30f; cs[c] = 0.f; }
    #pragma unroll
    for (int r = 0; r < 4; ++r) {
      int n = tid + 1024 * r;
      float rp[16];
      loadRow16(C + (size_t)n * 16, rp);
      float m = -1e30f;
      #pragma unroll
      for (int s = 0; s < 16; ++s) m = fmaxf(m, vcur[s] - rp[s]);
      float ss = 0.f;
      #pragma unroll
      for (int s = 0; s < 16; ++s) ss += __expf(vcur[s] - rp[s] - m);
      ur[r] = la[r] - (m + __logf(ss));
      #pragma unroll
      for (int s = 0; s < 16; ++s) {
        float tv = ur[r] - rp[s];
        float M = fmaxf(cm[s], tv);
        cs[s] = cs[s] * __expf(cm[s] - M) + __expf(tv - M);
        cm[s] = M;
      }
    }
    blockColLSE(cm, cs, Mt, St, redm, reds, tid);
    if (tid < 16) {
      float vn = lbs[tid] - (Mt[tid] + __logf(St[tid]));
      vcur[tid] = vn; vh[k][tid] = vn;
    }
    __syncthreads();
  }
  #pragma unroll
  for (int r = 0; r < 4; ++r) u[b * KN + tid + 1024 * r] = ur[r];
  if (tid < 16) v[b * 16 + tid] = vcur[tid];

  // ---- backward init: G = gA * attn, attn = exp(u5+v5-C)
  const float invNS = 1.f / ((float)KN * (float)KS);
  float gP[4][16];
  float gur[4];
  float cp[16];
  #pragma unroll
  for (int c = 0; c < 16; ++c) cp[c] = 0.f;
  #pragma unroll
  for (int r = 0; r < 4; ++r) {
    int n = tid + 1024 * r;
    float rp[16];
    loadRow16(C + (size_t)n * 16, rp);
    float grow = 0.f;
    #pragma unroll
    for (int s = 0; s < 16; ++s) {
      float at = __expf(ur[r] + vcur[s] - rp[s]);
      float G = 0.f;
      if (at > 1e-20f && at < 1.f) G = -(1.f + __logf(at)) * at * invNS;
      gP[r][s] = G; grow += G; cp[s] += G;
    }
    gur[r] = grow;
  }
  blockColSum(cp, gvs, redm, tid);

  // ---- backward through 5 iterations (v history only; u_k recomputed)
  for (int k = 5; k >= 1; --k) {
    #pragma unroll
    for (int c = 0; c < 16; ++c) cp[c] = 0.f;
    #pragma unroll
    for (int r = 0; r < 4; ++r) {
      float rp[16];
      int n = tid + 1024 * r;
      loadRow16(C + (size_t)n * 16, rp);
      float uk;
      if (k == 5) uk = ur[r];
      else {
        float m = -1e30f;
        #pragma unroll
        for (int s = 0; s < 16; ++s) m = fmaxf(m, vh[k - 1][s] - rp[s]);
        float ss = 0.f;
        #pragma unroll
        for (int s = 0; s < 16; ++s) ss += __expf(vh[k - 1][s] - rp[s] - m);
        uk = la[r] - (m + __logf(ss));
      }
      float rs = 0.f;
      #pragma unroll
      for (int s = 0; s < 16; ++s) {
        float tv = -gvs[s] * __expf(uk + vh[k][s] - rp[s] - lbs[s]);
        gP[r][s] += tv; rs += tv;
      }
      float gu_k = ((k == 5) ? gur[r] : 0.f) + rs;
      #pragma unroll
      for (int s = 0; s < 16; ++s) {
        float t2 = -gu_k * __expf(vh[k - 1][s] + uk - rp[s] - la[r]);
        gP[r][s] += t2; cp[s] += t2;
      }
    }
    blockColSum(cp, gvs, redm, tid);
  }

  // ---- g = -gP + 1e-20 ; per-batch L2 norm ; C_t -= 5*g/max(nrm,1e-12)
  float ssq = 0.f;
  #pragma unroll
  for (int r = 0; r < 4; ++r)
    #pragma unroll
    for (int s = 0; s < 16; ++s) {
      float gg = -gP[r][s] + 1e-20f;
      gP[r][s] = gg; ssq += gg * gg;
    }
  float tot = blockSum1024(ssq, red32, tid);
  float inv = 5.f / fmaxf(sqrtf(tot), 1e-12f);
  #pragma unroll
  for (int r = 0; r < 4; ++r) {
    int n = tid + 1024 * r;
    float rp[16];
    loadRow16(C + (size_t)n * 16, rp);
    #pragma unroll
    for (int s = 0; s < 16; ++s)
      C[(size_t)n * 16 + s] = rp[s] - inv * gP[r][s];
  }
}

// ---------------------------------------------------------------- final sinkhorn → attn (+ f16 transposed copy for WMMA updates)

__global__ void sinkhorn_attn_kernel(const float* __restrict__ Ct, const float* __restrict__ u,
                                     const float* __restrict__ v, const float* __restrict__ log_a,
                                     const float* __restrict__ log_b, float* __restrict__ attn,
                                     _Float16* __restrict__ attnT16) {
  __shared__ float redm[32][16], reds[32][16];
  __shared__ float vcur[16], lbs[16], Mt[16], St[16];
  int b = blockIdx.x, tid = threadIdx.x;
  const float* C = Ct + (size_t)b * KN * 16;
  float ur[4], la[4];
  #pragma unroll
  for (int r = 0; r < 4; ++r) {
    ur[r] = u[b * KN + tid + 1024 * r];
    la[r] = log_a[b * KN + tid + 1024 * r];
  }
  if (tid < 16) { vcur[tid] = v[b * 16 + tid]; lbs[tid] = log_b[b * 16 + tid]; }
  __syncthreads();
  for (int k = 1; k <= 5; ++k) {
    float cm[16], cs[16];
    #pragma unroll
    for (int c = 0; c < 16; ++c) { cm[c] = -1e30f; cs[c] = 0.f; }
    #pragma unroll
    for (int r = 0; r < 4; ++r) {
      int n = tid + 1024 * r;
      float rp[16];
      loadRow16(C + (size_t)n * 16, rp);
      float m = -1e30f;
      #pragma unroll
      for (int s = 0; s < 16; ++s) m = fmaxf(m, vcur[s] - rp[s]);
      float ss = 0.f;
      #pragma unroll
      for (int s = 0; s < 16; ++s) ss += __expf(vcur[s] - rp[s] - m);
      ur[r] = la[r] - (m + __logf(ss));
      #pragma unroll
      for (int s = 0; s < 16; ++s) {
        float tv = ur[r] - rp[s];
        float M = fmaxf(cm[s], tv);
        cs[s] = cs[s] * __expf(cm[s] - M) + __expf(tv - M);
        cm[s] = M;
      }
    }
    blockColLSE(cm, cs, Mt, St, redm, reds, tid);
    if (tid < 16) vcur[tid] = lbs[tid] - (Mt[tid] + __logf(St[tid]));
    __syncthreads();
  }
  #pragma unroll
  for (int r = 0; r < 4; ++r) {
    int n = tid + 1024 * r;
    float rp[16];
    loadRow16(C + (size_t)n * 16, rp);
    #pragma unroll
    for (int s = 0; s < 16; ++s) {
      float at = __expf(ur[r] + vcur[s] - rp[s]);
      attn[((size_t)(b * KN + n)) * 16 + s] = at;
      attnT16[((size_t)(b * 16 + s)) * KN + n] = (_Float16)at;
    }
  }
}

// ---------------------------------------------------------------- WMMA split-K updates: updates[b,s,d] = sum_n attn[b,n,s]*val[b,n,d]

__global__ void updates_wmma_kernel(const unsigned* __restrict__ attnT16w, // [b][s][2048 npairs]
                                    const unsigned* __restrict__ vpT,      // [b][d][2048 npairs]
                                    float* __restrict__ updates) {
  int lane = threadIdx.x & 31;
  int nt = blockIdx.x;    // d tile 0..7
  int kc = blockIdx.y;    // k slice 0..7
  int b  = blockIdx.z;
  int half = lane >> 4, mr = lane & 15;
  int n0 = nt * 16;
  const unsigned* arow = attnT16w + (size_t)(b * 16 + mr) * 2048;
  const unsigned* brow = vpT + (size_t)(b * 128 + n0 + mr) * 2048;
  v8f c = {0.f, 0.f, 0.f, 0.f, 0.f, 0.f, 0.f, 0.f};
  int kbase = kc * 512;
  for (int k0 = kbase; k0 < kbase + 512; k0 += 32) {
    __builtin_prefetch(arow + ((k0 + 32) >> 1), 0, 1);
    __builtin_prefetch(brow + ((k0 + 32) >> 1), 0, 1);
    union { v16h v; unsigned w[8]; } a, bb_;
    #pragma unroll
    for (int p = 0; p < 8; ++p) {
      int pa = (p < 4) ? (p + 4 * half) : (8 + (p - 4) + 4 * half);
      a.w[p] = arow[(k0 >> 1) + pa];
      bb_.w[p] = brow[(k0 >> 1) + pa];
    }
    c = __builtin_amdgcn_wmma_f32_16x16x32_f16(false, a.v, false, bb_.v, (short)0, c, false, false);
  }
  #pragma unroll
  for (int r = 0; r < 8; ++r) {
    int m = r + 8 * half;          // slot index
    atomicAdd(&updates[(b * 16 + m) * KDS + n0 + mr], c[r]);
  }
}

// ---------------------------------------------------------------- GRU / MLP

__global__ void gru_kernel(const float* __restrict__ updates, float* __restrict__ slots,
                           const float* __restrict__ wi, const float* __restrict__ wh,
                           const float* __restrict__ bi, const float* __restrict__ bh) {
  __shared__ float upd[128], sprev[128], gi[384], gh[384];
  int bs = blockIdx.x, t = threadIdx.x;
  if (t < 128) { upd[t] = updates[bs * KDS + t]; sprev[t] = slots[bs * KDS + t]; }
  __syncthreads();
  float ai = bi[t], ah = bh[t];
  for (int k = 0; k < 128; ++k) {
    ai += upd[k] * wi[t * KDS + k];
    ah += sprev[k] * wh[t * KDS + k];
  }
  gi[t] = ai; gh[t] = ah;
  __syncthreads();
  if (t < 128) {
    float r = 1.f / (1.f + __expf(-(gi[t] + gh[t])));
    float z = 1.f / (1.f + __expf(-(gi[128 + t] + gh[128 + t])));
    float nn = tanhf(gi[256 + t] + r * gh[256 + t]);
    slots[bs * KDS + t] = (1.f - z) * nn + z * sprev[t];
  }
}

__global__ void mlp_kernel(float* __restrict__ slots, const float* __restrict__ g,
                           const float* __restrict__ be, const float* __restrict__ W1,
                           const float* __restrict__ b1, const float* __restrict__ W2,
                           const float* __restrict__ b2) {
  __shared__ float sm[128], sorig[128], h[256], red[256];
  int bs = blockIdx.x, t = threadIdx.x;
  float x = 0.f;
  if (t < 128) { x = slots[bs * KDS + t]; sorig[t] = x; }
  red[t] = (t < 128) ? x : 0.f; __syncthreads();
  for (int o = 128; o >= 1; o >>= 1) { if (t < o) red[t] += red[t + o]; __syncthreads(); }
  float mean = red[0] * (1.f / 128.f); __syncthreads();
  float d = x - mean;
  red[t] = (t < 128) ? d * d : 0.f; __syncthreads();
  for (int o = 128; o >= 1; o >>= 1) { if (t < o) red[t] += red[t + o]; __syncthreads(); }
  float var = red[0] * (1.f / 128.f); __syncthreads();
  if (t < 128) sm[t] = d * rsqrtf(var + 1e-5f) * g[t] + be[t];
  __syncthreads();
  float hv = b1[t];
  for (int k = 0; k < 128; ++k) hv += sm[k] * W1[t * KDS + k];
  h[t] = fmaxf(hv, 0.f);
  __syncthreads();
  if (t < 128) {
    float o = b2[t];
    for (int j = 0; j < 256; ++j) o += h[j] * W2[t * KDH + j];
    slots[bs * KDS + t] = sorig[t] + o;
  }
}

__global__ void output_kernel(const float* __restrict__ slots, const float* __restrict__ attn,
                              float* __restrict__ out) {
  int i = blockIdx.x * 256 + threadIdx.x;
  int total = KB * KS * KDS + KB * KS * KN;
  if (i >= total) return;
  if (i < KB * KS * KDS) { out[i] = slots[i]; return; }
  int j = i - KB * KS * KDS;
  int b = j / (KS * KN);
  int rem = j % (KS * KN);
  int s = rem / KN, n = rem % KN;
  out[i] = attn[((size_t)(b * KN + n)) * 16 + s];
}

// ---------------------------------------------------------------- host launch

extern "C" void kernel_launch(void* const* d_in, const int* in_sizes, int n_in,
                              void* d_out, int out_size, void* d_ws, size_t ws_size,
                              hipStream_t stream) {
  const float* inputs     = (const float*)d_in[0];
  const float* slots_init = (const float*)d_in[1];
  const float* ln_in_g    = (const float*)d_in[2];
  const float* ln_in_b    = (const float*)d_in[3];
  const float* ln_sl_g    = (const float*)d_in[4];
  const float* ln_sl_b    = (const float*)d_in[5];
  const float* ln_mlp_g   = (const float*)d_in[6];
  const float* ln_mlp_b   = (const float*)d_in[7];
  const float* Wq = (const float*)d_in[8];
  const float* Wk = (const float*)d_in[9];
  const float* Wv = (const float*)d_in[10];
  const float* Wa = (const float*)d_in[11];
  const float* ba = (const float*)d_in[12];
  const float* Wb = (const float*)d_in[13];
  const float* bb = (const float*)d_in[14];
  const float* gru_wi = (const float*)d_in[15];
  const float* gru_wh = (const float*)d_in[16];
  const float* gru_bi = (const float*)d_in[17];
  const float* gru_bh = (const float*)d_in[18];
  const float* W1 = (const float*)d_in[19];
  const float* b1 = (const float*)d_in[20];
  const float* W2 = (const float*)d_in[21];
  const float* b2 = (const float*)d_in[22];

  char* base = (char*)d_ws;
  size_t off = 0;
  auto alloc = [&](size_t bytes) { size_t p = off; off += (bytes + 255) & ~(size_t)255; return p; };
  size_t o_x16   = alloc((size_t)KR * 256 * 2);
  size_t o_kv    = alloc((size_t)KR * 256 * 4);
  size_t o_kk    = alloc((size_t)KR * 4);
  size_t o_alog  = alloc((size_t)KR * 4);
  size_t o_loga  = alloc((size_t)KR * 4);
  size_t o_wp    = alloc(128 * 256 * 4);
  size_t o_slots = alloc(KB * KS * KDS * 4);
  size_t o_q     = alloc(KB * KS * KDS * 4);
  size_t o_qpack = alloc(KB * 64 * 16 * 4);
  size_t o_qq    = alloc(KB * 16 * 4);
  size_t o_blog  = alloc(KB * 16 * 4);
  size_t o_logb  = alloc(KB * 16 * 4);
  size_t o_u     = alloc((size_t)KR * 4);
  size_t o_v     = alloc(KB * 16 * 4);
  size_t o_Ct    = alloc((size_t)KR * 16 * 4);
  size_t o_attn  = alloc((size_t)KR * 16 * 4);
  size_t o_at16  = alloc((size_t)KR * 16 * 2);
  size_t o_vpt   = alloc((size_t)KB * 128 * 2048 * 4);
  size_t o_upd   = alloc(KB * KS * KDS * 4);
  (void)ws_size; (void)n_in; (void)in_sizes; (void)out_size;

  _Float16* x16   = (_Float16*)(base + o_x16);
  float* kv       = (float*)(base + o_kv);
  float* kk       = (float*)(base + o_kk);
  float* alogit   = (float*)(base + o_alog);
  float* log_a    = (float*)(base + o_loga);
  unsigned* wpT   = (unsigned*)(base + o_wp);
  float* slots    = (float*)(base + o_slots);
  float* q        = (float*)(base + o_q);
  unsigned* qpackT= (unsigned*)(base + o_qpack);
  float* qq       = (float*)(base + o_qq);
  float* blogit   = (float*)(base + o_blog);
  float* log_b    = (float*)(base + o_logb);
  float* u        = (float*)(base + o_u);
  float* v        = (float*)(base + o_v);
  float* Ct       = (float*)(base + o_Ct);
  float* attn     = (float*)(base + o_attn);
  _Float16* at16  = (_Float16*)(base + o_at16);
  unsigned* vpT   = (unsigned*)(base + o_vpt);
  float* updates  = (float*)(base + o_upd);
  float* out      = (float*)d_out;

  pack_w_kernel<<<dim3(128), 256, 0, stream>>>(Wk, Wv, wpT);
  prep_ln_kernel<<<KR, 256, 0, stream>>>(inputs, ln_in_g, ln_in_b, Wa, ba, x16, alogit);
  gemm_xw_kernel<<<dim3(KR / 16, 2), 256, 0, stream>>>((const unsigned*)x16, wpT, kv);
  rownorm_kernel<<<KR / 256, 256, 0, stream>>>(kv, kk);
  pack_val_kernel<<<(KB * 128 * 2048) / 256, 256, 0, stream>>>(kv, vpT);
  softmax_loga_kernel<<<KB, 1024, 0, stream>>>(alogit, log_a);
  copy_kernel<<<(KB * KS * KDS + 255) / 256, 256, 0, stream>>>(slots_init, slots, KB * KS * KDS);

  for (int i = 0; i < 3; ++i) {
    slot_ln_kernel<<<KB * KS, 128, 0, stream>>>(slots, ln_sl_g, ln_sl_b, Wb, bb, Wq, q, blogit, qq);
    slot_small_kernel<<<KB, 1024, 0, stream>>>(blogit, q, log_b, qpackT);
    cdist_kernel<<<dim3(KN / 16, KB), 32, 0, stream>>>(kv, qpackT, kk, qq, Ct, i);
    zero_uv_kernel<<<KR / 256, 256, 0, stream>>>(u, v);
    for (int t = 0; t < 4; ++t)
      mesh_step_kernel<<<KB, 1024, 0, stream>>>(Ct, u, v, log_a, log_b);
    sinkhorn_attn_kernel<<<KB, 1024, 0, stream>>>(Ct, u, v, log_a, log_b, attn, at16);
    zero_kernel<<<(KB * KS * KDS + 255) / 256, 256, 0, stream>>>(updates, KB * KS * KDS);
    updates_wmma_kernel<<<dim3(8, 8, KB), 32, 0, stream>>>((const unsigned*)at16,
                                                           vpT, updates);
    gru_kernel<<<KB * KS, 384, 0, stream>>>(updates, slots, gru_wi, gru_wh, gru_bi, gru_bh);
    mlp_kernel<<<KB * KS, 256, 0, stream>>>(slots, ln_mlp_g, ln_mlp_b, W1, b1, W2, b2);
  }

  int total = KB * KS * KDS + KB * KS * KN;
  output_kernel<<<(total + 255) / 256, 256, 0, stream>>>(slots, attn, out);
}
